// QFormerBlock_3607772528675
// MI455X (gfx1250) — matrix operand order repaired and checked by reading, not compiled, not run on hardware
//
#include <hip/hip_runtime.h>

typedef __attribute__((ext_vector_type(16))) __bf16 v16bf;
typedef __attribute__((ext_vector_type(8)))  float  v8f;

__device__ inline __bf16 f2bf(float x) {
    unsigned u = __builtin_bit_cast(unsigned, x);
    u += 0x7FFFu + ((u >> 16) & 1u);
    unsigned short h = (unsigned short)(u >> 16);
    return __builtin_bit_cast(__bf16, h);
}

__device__ inline v8f vzero8() {
    v8f z;
#pragma unroll
    for (int i = 0; i < 8; ++i) z[i] = 0.f;
    return z;
}

// ---------------------------------------------------------------------------
// Tiled bf16 WMMA GEMM: C[M,N] = A[M,K] @ B[K,N] + bias, all f32 in memory.
// Block tile 128x128, K-step 32. 256 threads = 8 waves; wave grid 4(M) x 2(N),
// each wave computes 32x64 via 2x4 fragments of 16x16.
// ---------------------------------------------------------------------------
__global__ __launch_bounds__(256)
void qf_gemm(const float* __restrict__ A, const float* __restrict__ B,
             const float* __restrict__ bias, float* __restrict__ C,
             int M, int N, int K)
{
    __shared__ __bf16 As[128][40];   // [m][k], padded
    __shared__ __bf16 Bs[128][40];   // transposed [n][k], padded

    const int tid  = threadIdx.x;
    const int lane = tid & 31;
    const int wave = tid >> 5;
    const int wm   = wave & 3;      // M sub-tile (32 rows)
    const int wn   = wave >> 2;     // N sub-tile (64 cols)
    const int ln   = lane & 15;
    const int hi   = lane >> 4;
    const int m0   = blockIdx.y * 128;
    const int n0   = blockIdx.x * 128;

    v8f acc[2][4];
#pragma unroll
    for (int i = 0; i < 2; ++i)
#pragma unroll
        for (int j = 0; j < 4; ++j) acc[i][j] = vzero8();

    for (int k0 = 0; k0 < K; k0 += 32) {
        __syncthreads();
        {   // stage A 128x32 (coalesced along k)
            int r  = tid >> 3;
            int kc = (tid & 7) * 4;
#pragma unroll
            for (int p = 0; p < 4; ++p) {
                int row = r + p * 32;
                const float4 v = *(const float4*)(&A[(size_t)(m0 + row) * K + k0 + kc]);
                As[row][kc + 0] = f2bf(v.x); As[row][kc + 1] = f2bf(v.y);
                As[row][kc + 2] = f2bf(v.z); As[row][kc + 3] = f2bf(v.w);
            }
        }
        {   // stage B 32x128 -> transposed LDS [n][k] (coalesced along n)
            int kr = tid >> 5;
            int nc = (tid & 31) * 4;
#pragma unroll
            for (int p = 0; p < 4; ++p) {
                int kk = kr + p * 8;
                const float4 v = *(const float4*)(&B[(size_t)(k0 + kk) * N + n0 + nc]);
                Bs[nc + 0][kk] = f2bf(v.x); Bs[nc + 1][kk] = f2bf(v.y);
                Bs[nc + 2][kk] = f2bf(v.z); Bs[nc + 3][kk] = f2bf(v.w);
            }
        }
        __syncthreads();

        v16bf af[2], bf[4];
#pragma unroll
        for (int am = 0; am < 2; ++am) {
            const __bf16* ap = &As[wm * 32 + am * 16 + ln][hi * 16];
#pragma unroll
            for (int i = 0; i < 8; ++i) { af[am][i] = ap[i]; af[am][8 + i] = ap[8 + i]; }
        }
#pragma unroll
        for (int bn = 0; bn < 4; ++bn) {
            const __bf16* bp = &Bs[wn * 64 + bn * 16 + ln][hi * 16];
#pragma unroll
            for (int i = 0; i < 16; ++i) bf[bn][i] = bp[i];
        }
#pragma unroll
        for (int am = 0; am < 2; ++am)
#pragma unroll
            for (int bn = 0; bn < 4; ++bn)
                acc[am][bn] = __builtin_amdgcn_wmma_f32_16x16x32_bf16(
                    false, af[am], false, bf[bn], (short)0, acc[am][bn], false, false);
    }

#pragma unroll
    for (int am = 0; am < 2; ++am) {
        int mrow = m0 + wm * 32 + am * 16 + hi * 8;
#pragma unroll
        for (int bn = 0; bn < 4; ++bn) {
            int col = n0 + wn * 64 + bn * 16 + ln;
            float bv = bias ? bias[col] : 0.f;
#pragma unroll
            for (int r = 0; r < 8; ++r)
                C[(size_t)(mrow + r) * N + col] = acc[am][bn][r] + bv;
        }
    }
}

// ---------------------------------------------------------------------------
// Flash attention over QKV buffer [4096, 3072] (per row: 16 heads x (q64,k64,v64)).
// Grid: x = query block (128 rows) of S=1024, y = b*16+h. 256 threads = 8 waves,
// each wave owns 16 query rows. Online softmax, j-steps of 32.
// Output: Ctx[4096, 1024] with head h at columns h*64..h*64+63.
// ---------------------------------------------------------------------------
__global__ __launch_bounds__(256)
void qf_flash(const float* __restrict__ QKV, float* __restrict__ Ctx)
{
    const int mb   = blockIdx.x;
    const int bh   = blockIdx.y;
    const int b    = bh >> 4, h = bh & 15;
    const int tid  = threadIdx.x;
    const int lane = tid & 31, wave = tid >> 5;
    const int ln   = lane & 15, hi = lane >> 4;

    __shared__ __bf16 Qs[128][72];
    __shared__ __bf16 Ks[32][72];
    __shared__ __bf16 Vs[64][40];     // transposed [d][j]
    __shared__ __bf16 Ps[8][16][40];  // per-wave P staging

    const size_t rowbase = (size_t)b * 1024;
    const int qoff = h * 192;

    {   // stage Q 128x64
        int d = (tid & 15) * 4;
        int r = tid >> 4;
#pragma unroll
        for (int p = 0; p < 8; ++p) {
            int row = r + p * 16;
            const float4 v = *(const float4*)(&QKV[(rowbase + mb * 128 + row) * 3072 + qoff + d]);
            Qs[row][d + 0] = f2bf(v.x); Qs[row][d + 1] = f2bf(v.y);
            Qs[row][d + 2] = f2bf(v.z); Qs[row][d + 3] = f2bf(v.w);
        }
    }

    v8f O[4];
#pragma unroll
    for (int i = 0; i < 4; ++i) O[i] = vzero8();
    float Mrow[8], Lrow[8];
#pragma unroll
    for (int r = 0; r < 8; ++r) { Mrow[r] = -3.0e38f; Lrow[r] = 0.f; }

    for (int j0 = 0; j0 < 1024; j0 += 32) {
        __syncthreads();
        {   // stage K 32x64
            int d = (tid & 15) * 4;
            int r = tid >> 4;
#pragma unroll
            for (int p = 0; p < 2; ++p) {
                int row = r + p * 16;
                const float4 v = *(const float4*)(&QKV[(rowbase + j0 + row) * 3072 + qoff + 64 + d]);
                Ks[row][d + 0] = f2bf(v.x); Ks[row][d + 1] = f2bf(v.y);
                Ks[row][d + 2] = f2bf(v.z); Ks[row][d + 3] = f2bf(v.w);
            }
        }
        {   // stage V 32x64 transposed -> [d][j]
            int d = (tid & 15) * 4;
            int r = tid >> 4;
#pragma unroll
            for (int p = 0; p < 2; ++p) {
                int row = r + p * 16;
                const float4 v = *(const float4*)(&QKV[(rowbase + j0 + row) * 3072 + qoff + 128 + d]);
                Vs[d + 0][row] = f2bf(v.x); Vs[d + 1][row] = f2bf(v.y);
                Vs[d + 2][row] = f2bf(v.z); Vs[d + 3][row] = f2bf(v.w);
            }
        }
        __syncthreads();

        // logits S[2] = Q(16x64) @ K^T(64x32), f32 accum
        v8f S[2];
#pragma unroll
        for (int jt = 0; jt < 2; ++jt) {
            v8f s = vzero8();
#pragma unroll
            for (int ks = 0; ks < 2; ++ks) {
                v16bf qa, kb;
                const __bf16* qp = &Qs[wave * 16 + ln][ks * 32 + hi * 16];
#pragma unroll
                for (int i = 0; i < 8; ++i) { qa[i] = qp[i]; qa[8 + i] = qp[8 + i]; }
                const __bf16* kp = &Ks[jt * 16 + ln][ks * 32 + hi * 16];
#pragma unroll
                for (int i = 0; i < 16; ++i) kb[i] = kp[i];
                s = __builtin_amdgcn_wmma_f32_16x16x32_bf16(false, qa, false, kb, (short)0, s, false, false);
            }
#pragma unroll
            for (int r = 0; r < 8; ++r) s[r] *= 0.125f;   // 1/sqrt(64)
            S[jt] = s;
        }

        // online softmax per row (row = r + 8*hi, cols striped over 16 lanes)
        float alpha[8];
#pragma unroll
        for (int r = 0; r < 8; ++r) {
            float mx = fmaxf(S[0][r], S[1][r]);
#pragma unroll
            for (int off = 1; off < 16; off <<= 1) mx = fmaxf(mx, __shfl_xor(mx, off, 32));
            float newM = fmaxf(Mrow[r], mx);
            alpha[r] = __expf(Mrow[r] - newM);
            Mrow[r] = newM;
            float p0 = __expf(S[0][r] - newM);
            float p1 = __expf(S[1][r] - newM);
            S[0][r] = p0; S[1][r] = p1;
            float ls = p0 + p1;
#pragma unroll
            for (int off = 1; off < 16; off <<= 1) ls += __shfl_xor(ls, off, 32);
            Lrow[r] = Lrow[r] * alpha[r] + ls;
        }
#pragma unroll
        for (int nd = 0; nd < 4; ++nd)
#pragma unroll
            for (int r = 0; r < 8; ++r) O[nd][r] *= alpha[r];

        // P: C-frag layout -> LDS -> A-frag layout (per-wave private, in-order LDS)
#pragma unroll
        for (int r = 0; r < 8; ++r) {
            Ps[wave][r + hi * 8][ln]      = f2bf(S[0][r]);
            Ps[wave][r + hi * 8][16 + ln] = f2bf(S[1][r]);
        }
        v16bf pa;
        const __bf16* pp = &Ps[wave][ln][hi * 16];
#pragma unroll
        for (int i = 0; i < 8; ++i) { pa[i] = pp[i]; pa[8 + i] = pp[8 + i]; }

#pragma unroll
        for (int nd = 0; nd < 4; ++nd) {
            v16bf vb;
            const __bf16* vp = &Vs[nd * 16 + ln][hi * 16];
#pragma unroll
            for (int i = 0; i < 16; ++i) vb[i] = vp[i];
            O[nd] = __builtin_amdgcn_wmma_f32_16x16x32_bf16(false, pa, false, vb, (short)0, O[nd], false, false);
        }
    }

#pragma unroll
    for (int r = 0; r < 8; ++r) {
        float inv = 1.f / Lrow[r];
#pragma unroll
        for (int nd = 0; nd < 4; ++nd) O[nd][r] *= inv;
    }
#pragma unroll
    for (int nd = 0; nd < 4; ++nd) {
        int col = h * 64 + nd * 16 + ln;
#pragma unroll
        for (int r = 0; r < 8; ++r) {
            int row = mb * 128 + wave * 16 + r + hi * 8;
            Ctx[(rowbase + row) * 1024 + col] = O[nd][r];
        }
    }
}

// ---------------------------------------------------------------------------
// Batched (B=4) k-split GEMV: P[(s*4+b)*N+n] = sum_{k in slice s} X[b,K]·W[K,N]
// ---------------------------------------------------------------------------
#define KSPLIT 8
__global__ __launch_bounds__(256)
void qf_gemv4_part(const float* __restrict__ X, const float* __restrict__ W,
                   float* __restrict__ P, int K, int N)
{
    int n = blockIdx.x * 256 + threadIdx.x;
    if (n >= N) return;
    int k0 = blockIdx.y * (K / KSPLIT), k1 = k0 + K / KSPLIT;
    float a0 = 0.f, a1 = 0.f, a2 = 0.f, a3 = 0.f;
    for (int k = k0; k < k1; ++k) {
        float w = W[(size_t)k * N + n];
        a0 += X[k] * w; a1 += X[K + k] * w; a2 += X[2 * K + k] * w; a3 += X[3 * K + k] * w;
    }
    size_t base = (size_t)blockIdx.y * 4 * N + n;
    P[base] = a0; P[base + N] = a1; P[base + 2 * (size_t)N] = a2; P[base + 3 * (size_t)N] = a3;
}

__global__ __launch_bounds__(256)
void qf_gemv4_reduce(const float* __restrict__ P, const float* __restrict__ bias,
                     float* __restrict__ Y, int N, int act)
{
    int idx = blockIdx.x * 256 + threadIdx.x;   // over 4*N
    if (idx >= 4 * N) return;
    int b = idx / N, n = idx - b * N;
    float s = 0.f;
#pragma unroll
    for (int k = 0; k < KSPLIT; ++k) s += P[((size_t)k * 4 + b) * N + n];
    if (bias) s += bias[n];
    if (act) s = tanhf(s);
    Y[idx] = s;
}

// g[b,h,d] = sum_c ctx_qk_w[d, h*1024+c] * qk[b, h*1024+c]
__global__ __launch_bounds__(256)
void qf_cross_g(const float* __restrict__ Wqk, const float* __restrict__ qk,
                float* __restrict__ g)
{
    int idx = blockIdx.x * 256 + threadIdx.x;   // 16384
    int h = idx >> 10, d = idx & 1023;
    const float* wrow = Wqk + (size_t)d * 16384 + h * 1024;
    const float* q0 = qk + h * 1024;
    float a0 = 0.f, a1 = 0.f, a2 = 0.f, a3 = 0.f;
    for (int c = 0; c < 1024; ++c) {
        float w = wrow[c];
        a0 += w * q0[c]; a1 += w * q0[16384 + c]; a2 += w * q0[32768 + c]; a3 += w * q0[49152 + c];
    }
    g[(0 * 16 + h) * 1024 + d] = a0; g[(16 + h) * 1024 + d] = a1;
    g[(32 + h) * 1024 + d] = a2;     g[(48 + h) * 1024 + d] = a3;
}

// sim[b,h,j] = scale * dot(text1[b,j,:], g[b,h,:]); one wave per (b,j)
__global__ __launch_bounds__(256)
void qf_cross_sim(const float* __restrict__ T1, const float* __restrict__ g,
                  float* __restrict__ sim)
{
    int w = blockIdx.x * 8 + (threadIdx.x >> 5);
    int lane = threadIdx.x & 31;
    int b = w >> 10, j = w & 1023;
    const float* row = T1 + (size_t)(b * 1024 + j) * 1024;
    float x[32];
#pragma unroll
    for (int i = 0; i < 32; ++i) x[i] = row[lane + 32 * i];
    for (int h = 0; h < 16; ++h) {
        const float* gh = g + (size_t)(b * 16 + h) * 1024;
        float s = 0.f;
#pragma unroll
        for (int i = 0; i < 32; ++i) s += x[i] * gh[lane + 32 * i];
#pragma unroll
        for (int off = 16; off > 0; off >>= 1) s += __shfl_xor(s, off, 32);
        if (lane == 0) sim[(size_t)(b * 16 + h) * 1024 + j] = s * 0.03125f;  // 1024^-0.5
    }
}

__global__ __launch_bounds__(256)
void qf_softmax(float* __restrict__ sim)
{
    int t = threadIdx.x;
    float* row = sim + (size_t)blockIdx.x * 1024;
    __shared__ float red[256];
    float v[4]; float mx = -3.0e38f;
#pragma unroll
    for (int i = 0; i < 4; ++i) { v[i] = row[t + 256 * i]; mx = fmaxf(mx, v[i]); }
    red[t] = mx; __syncthreads();
    for (int s = 128; s > 0; s >>= 1) { if (t < s) red[t] = fmaxf(red[t], red[t + s]); __syncthreads(); }
    mx = red[0]; __syncthreads();
    float sum = 0.f;
#pragma unroll
    for (int i = 0; i < 4; ++i) { v[i] = __expf(v[i] - mx); sum += v[i]; }
    red[t] = sum; __syncthreads();
    for (int s = 128; s > 0; s >>= 1) { if (t < s) red[t] += red[t + s]; __syncthreads(); }
    float inv = 1.f / red[0];
#pragma unroll
    for (int i = 0; i < 4; ++i) row[t + 256 * i] = v[i] * inv;
}

// tw[b,h,:] = attn[b,h,:] @ text1[b]; grid (b, dchunk of 64); thread = (h, 4 d's)
__global__ __launch_bounds__(256)
void qf_cross_tw(const float* __restrict__ attn, const float* __restrict__ T1,
                 float* __restrict__ tw)
{
    int b = blockIdx.x, dc = blockIdx.y;
    int h = threadIdx.x >> 4, dl = threadIdx.x & 15;
    int d = dc * 64 + dl * 4;
    const float* a = attn + (size_t)(b * 16 + h) * 1024;
    const float* Tb = T1 + (size_t)b * 1024 * 1024 + d;
    float acc[4] = {0.f, 0.f, 0.f, 0.f};
    for (int j = 0; j < 1024; ++j) {
        float aj = a[j];
        const float* r = Tb + (size_t)j * 1024;
#pragma unroll
        for (int i = 0; i < 4; ++i) acc[i] += aj * r[i];
    }
#pragma unroll
    for (int i = 0; i < 4; ++i) tw[(size_t)(b * 16 + h) * 1024 + d + i] = acc[i];
}

// out_pre[b, n] = dot(tw[b, n>>10, :], ctx_v_w[:, n])
__global__ __launch_bounds__(256)
void qf_cross_outpre(const float* __restrict__ tw, const float* __restrict__ Wv,
                     float* __restrict__ op)
{
    int idx = blockIdx.x * 256 + threadIdx.x;   // 4*16384
    int b = idx >> 14, n = idx & 16383;
    int h = n >> 10;
    const float* x = tw + (size_t)(b * 16 + h) * 1024;
    float s = 0.f;
    for (int k = 0; k < 1024; ++k) s += x[k] * Wv[(size_t)k * 16384 + n];
    op[idx] = s;
}

__global__ __launch_bounds__(256)
void qf_bcast(const float* __restrict__ t, float* __restrict__ out)
{
    int idx = blockIdx.x * 256 + threadIdx.x;   // 4*1024*1024
    int b = idx >> 20, d = idx & 1023;
    out[idx] = t[b * 1024 + d];
}

// ---------------------------------------------------------------------------
extern "C" void kernel_launch(void* const* d_in, const int* in_sizes, int n_in,
                              void* d_out, int out_size, void* d_ws, size_t ws_size,
                              hipStream_t stream)
{
    (void)in_sizes; (void)n_in; (void)out_size; (void)ws_size;
    const float* img   = (const float*)d_in[0];
    const float* text  = (const float*)d_in[1];
    const float* qkv_w = (const float*)d_in[2];
    const float* qkv_b = (const float*)d_in[3];
    const float* o_w   = (const float*)d_in[4];
    const float* o_b   = (const float*)d_in[5];
    const float* qk_w  = (const float*)d_in[6];
    const float* cqk_w = (const float*)d_in[7];
    const float* v_w   = (const float*)d_in[8];
    const float* cv_w  = (const float*)d_in[9];
    const float* out_w = (const float*)d_in[10];
    const float* out_b = (const float*)d_in[11];
    const float* co_w  = (const float*)d_in[12];
    const float* co_b  = (const float*)d_in[13];
    const float* w1 = (const float*)d_in[14];
    const float* b1 = (const float*)d_in[15];
    const float* w2 = (const float*)d_in[16];
    const float* b2 = (const float*)d_in[17];
    const float* w3 = (const float*)d_in[18];
    const float* b3 = (const float*)d_in[19];
    float* out = (float*)d_out;

    float* qkv   = (float*)d_ws;                         // 4096*3072
    float* ctx   = qkv + (size_t)4096 * 3072;            // 4096*1024
    float* text1 = ctx + (size_t)4096 * 1024;            // 4096*1024
    float* qk    = text1 + (size_t)4096 * 1024;          // 65536
    float* vv    = qk + 65536;                           // 65536
    float* g     = vv + 65536;                           // 65536
    float* sim   = g + 65536;                            // 65536 (attn in place)
    float* tw    = sim + 65536;                          // 65536
    float* op    = tw + 65536;                           // 65536
    float* yctx  = op + 65536;                           // 4096
    float* h1    = yctx + 4096;                          // 8192
    float* h2    = h1 + 8192;                            // 8192
    float* tm    = h2 + 8192;                            // 4096
    float* part  = tm + 4096;                            // KSPLIT*4*16384

    // 1. QKV = text @ qkv_w + qkv_b   [4096,3072]
    qf_gemm<<<dim3(24, 32), 256, 0, stream>>>(text, qkv_w, qkv_b, qkv, 4096, 3072, 1024);
    // 2. flash attention -> ctx [4096,1024]
    qf_flash<<<dim3(8, 64), 256, 0, stream>>>(qkv, ctx);
    // 3. text1 = ctx @ o_w + o_b  [4096,1024]
    qf_gemm<<<dim3(8, 32), 256, 0, stream>>>(ctx, o_w, o_b, text1, 4096, 1024, 1024);

    // 4. qk = img @ qk_w ; vv = img @ v_w   [4,16384]
    qf_gemv4_part<<<dim3(64, KSPLIT), 256, 0, stream>>>(img, qk_w, part, 1024, 16384);
    qf_gemv4_reduce<<<256, 256, 0, stream>>>(part, nullptr, qk, 16384, 0);
    qf_gemv4_part<<<dim3(64, KSPLIT), 256, 0, stream>>>(img, v_w, part, 1024, 16384);
    qf_gemv4_reduce<<<256, 256, 0, stream>>>(part, nullptr, vv, 16384, 0);

    // 5. cross-attention (algebraically collapsed)
    qf_cross_g<<<64, 256, 0, stream>>>(cqk_w, qk, g);
    qf_cross_sim<<<512, 256, 0, stream>>>(text1, g, sim);
    qf_softmax<<<64, 256, 0, stream>>>(sim);
    qf_cross_tw<<<dim3(4, 16), 256, 0, stream>>>(sim, text1, tw);
    qf_cross_outpre<<<256, 256, 0, stream>>>(tw, cv_w, op);

    // 6. out = op @ out_w + out_b  -> d_out[0:4096]
    qf_gemv4_part<<<dim3(4, KSPLIT), 256, 0, stream>>>(op, out_w, part, 16384, 1024);
    qf_gemv4_reduce<<<16, 256, 0, stream>>>(part, out_b, out, 1024, 0);
    // 7. yctx = vv @ ctx_out_w + ctx_out_b (row constant over S)
    qf_gemv4_part<<<dim3(4, KSPLIT), 256, 0, stream>>>(vv, co_w, part, 16384, 1024);
    qf_gemv4_reduce<<<16, 256, 0, stream>>>(part, co_b, yctx, 1024, 0);

    // 8. MLP on 4 rows
    qf_gemv4_part<<<dim3(8, KSPLIT), 256, 0, stream>>>(yctx, w1, part, 1024, 2048);
    qf_gemv4_reduce<<<32, 256, 0, stream>>>(part, b1, h1, 2048, 1);
    qf_gemv4_part<<<dim3(8, KSPLIT), 256, 0, stream>>>(h1, w2, part, 2048, 2048);
    qf_gemv4_reduce<<<32, 256, 0, stream>>>(part, b2, h2, 2048, 1);
    qf_gemv4_part<<<dim3(4, KSPLIT), 256, 0, stream>>>(h2, w3, part, 2048, 1024);
    qf_gemv4_reduce<<<16, 256, 0, stream>>>(part, b3, tm, 1024, 0);

    // 9. broadcast text_out over S -> d_out[4096:]
    qf_bcast<<<16384, 256, 0, stream>>>(tm, out + 4096);
}
